// DiehlCookNetwork_39754217292114
// MI455X (gfx1250) — matrix operand order repaired and checked by reading, not compiled, not run on hardware
//
#include <hip/hip_runtime.h>

// ---------------------------------------------------------------------------
// Diehl-Cook SNN scan for MI455X (gfx1250, wave32, WMMA, TDM).
//
// 8 persistent workgroups (one per 16-batch-row M tile) x 512 threads
// (16 waves). LIF state lives in VGPRs in WMMA C/D layout for all 500 steps.
// Both matmuls (x_t @ relu(w_in) and s_inh @ -relu(w_ie)) chain through f32
// accumulators via f16 WMMA 16x16x32. Per k-tile, the A fragment is loaded
// once and reused for this wave's two N-tiles (independent accumulators ->
// back-to-back WMMAs), and fragments are software-pipelined one k-tile ahead.
// Weight base pointers are rebound through an empty asm each step so LICM
// cannot hoist the (loop-invariant) weight fragments out of the 500-step
// loop and spill them to scratch -- they stream from L2 every step instead.
// The per-step activation slab (16x800 f16 = 25.6 KB contiguous) is
// double-buffered in LDS and fetched by the Tensor Data Mover one step ahead.
// ---------------------------------------------------------------------------

typedef __attribute__((ext_vector_type(16))) _Float16 v16h;
typedef __attribute__((ext_vector_type(8)))  _Float16 v8h;
typedef __attribute__((ext_vector_type(8)))  float    v8f;
typedef unsigned int u32;
typedef __attribute__((ext_vector_type(4))) u32 v4u;
typedef __attribute__((ext_vector_type(8))) int v8i;
typedef __attribute__((ext_vector_type(4))) int v4i;

typedef union { v16h v; v8h h[2]; } v16h_u;

#define BATCH   128
#define N_IN    784
#define K1PAD   800          // N_IN padded to multiple of 32
#define N_EXC   400
#define K2PAD   416          // N_EXC padded to multiple of 32
#define TSTEPS  500
#define NTILES  25           // N_EXC / 16
#define DECAY   0.99f        // 1 - DT/TAU
#define WEI     50.0f        // WEIGHT_EI
#define SLAB    (16 * K1PAD) // f16 elements per per-step activation slab

#if __has_builtin(__builtin_amdgcn_tensor_load_to_lds) && __has_builtin(__builtin_amdgcn_s_wait_tensorcnt)
#define HAVE_TDM 1
#else
#define HAVE_TDM 0
#endif

#define WMMA_F16(A, B, C) \
    __builtin_amdgcn_wmma_f32_16x16x32_f16(false, (A), false, (B), (short)0, (C), false, false)

// ---------------------------------------------------------------------------
// Prep kernels: relu/mask/negate + f16 + B-fragment tile packing, and
// (B,N,T)->(T,B,K1PAD) f16 transpose of the spike train.
// ---------------------------------------------------------------------------
__global__ void pack_w_in(const float* __restrict__ w, _Float16* __restrict__ pw) {
    int tid = blockIdx.x * blockDim.x + threadIdx.x;
    if (tid >= NTILES * K1PAD * 16) return;
    int s  = tid & 15;
    int r  = (tid >> 4) % K1PAD;
    int j  = r & 15;
    int k  = (r & ~31) + ((s & 8) << 1) + ((r >> 4) & 1) * 8 + (s & 7);
    int nt = tid / (16 * K1PAD);
    float v = 0.0f;
    if (k < N_IN) {
        float w0 = w[(size_t)k * N_EXC + nt * 16 + j];
        v = w0 > 0.0f ? w0 : 0.0f;
    }
    pw[tid] = (_Float16)v;
}

__global__ void pack_w_ie(const float* __restrict__ w, _Float16* __restrict__ pw) {
    int tid = blockIdx.x * blockDim.x + threadIdx.x;
    if (tid >= NTILES * K2PAD * 16) return;
    int s  = tid & 15;
    int r  = (tid >> 4) % K2PAD;
    int j  = r & 15;
    int k  = (r & ~31) + ((s & 8) << 1) + ((r >> 4) & 1) * 8 + (s & 7);
    int nt = tid / (16 * K2PAD);
    int n  = nt * 16 + j;
    float v = 0.0f;
    if (k < N_EXC && k != n) {
        float w0 = w[(size_t)k * N_EXC + n];
        v = w0 > 0.0f ? -w0 : 0.0f;
    }
    pw[tid] = (_Float16)v;
}

__global__ void transpose_x(const float* __restrict__ xin, _Float16* __restrict__ xT) {
    int tid = blockIdx.x * blockDim.x + threadIdx.x;
    if (tid >= BATCH * K1PAD) return;
    int k = tid % K1PAD;
    int b = tid / K1PAD;
    if (k < N_IN) {
        const float* src = xin + ((size_t)b * N_IN + k) * TSTEPS;
        for (int t = 0; t < TSTEPS; ++t)
            xT[((size_t)t * BATCH + b) * K1PAD + k] = (_Float16)src[t];
    } else {
        for (int t = 0; t < TSTEPS; ++t)
            xT[((size_t)t * BATCH + b) * K1PAD + k] = (_Float16)0.0f;
    }
}

// ---------------------------------------------------------------------------
// TDM: DMA one contiguous activation slab (12800 f16) from global to LDS.
// D# per CDNA5 ISA. 6-argument builtin form:
//   (uint32x4 g0, int32x8 g1, int32x4 g2, int32x4 g3, int32x8 g4, i32 cpol)
// ---------------------------------------------------------------------------
#if HAVE_TDM
__device__ __forceinline__ void tdm_load_slab(const _Float16* gsrc, u32 lds_off) {
    unsigned long long ga = (unsigned long long)(uintptr_t)gsrc;
    v4u g0;
    g0.x = 1u;                                             // count=1, user desc
    g0.y = lds_off;                                        // lds_addr (bytes)
    g0.z = (u32)(ga & 0xFFFFFFFFu);                        // global_addr[31:0]
    g0.w = (u32)((ga >> 32) & 0x01FFFFFFu) | (2u << 30);   // addr[56:32] | type=2
    v8i g1;
    g1[0] = (int)(1u << 16);                               // data_size = 2 bytes
    g1[1] = (int)((SLAB & 0xFFFFu) << 16);                 // tensor_dim0[15:0]
    g1[2] = (int)((SLAB >> 16) | (1u << 16));              // tensor_dim0[31:16], tensor_dim1=1
    g1[3] = (int)((SLAB & 0xFFFFu) << 16);                 // tensor_dim1 hi=0, tile_dim0=SLAB
    g1[4] = 1;                                             // tile_dim1=1, tile_dim2=0
    g1[5] = (int)SLAB;                                     // tensor_dim0_stride lo
    g1[6] = 0;
    g1[7] = 0;
    v4i g2 = {0, 0, 0, 0};
    v4i g3 = {0, 0, 0, 0};
    v8i g4 = {0, 0, 0, 0, 0, 0, 0, 0};
    __builtin_amdgcn_tensor_load_to_lds(g0, g1, g2, g3, g4, 0);
}
#endif

__device__ __forceinline__ void copy_slab(const _Float16* gsrc, _Float16* ldst,
                                          int tid, int nthreads) {
    const float4* s = (const float4*)gsrc;
    float4* d = (float4*)ldst;
    for (int c = tid; c < SLAB / 8; c += nthreads) d[c] = s[c];
}

// ---------------------------------------------------------------------------
// Main persistent kernel: one WG per 16 batch rows, 16 waves; wave w owns
// N-tiles {w, w+16 (if <25)}. All 500 steps run inside the kernel.
// ---------------------------------------------------------------------------
__global__ __launch_bounds__(512) void snn_scan(
    const _Float16* __restrict__ xT,     // (T, 128, 800) f16
    const _Float16* __restrict__ pwin,   // packed relu(w_in) tiles
    const _Float16* __restrict__ pwie,   // packed -relu(w_ie) tiles
    float* __restrict__ out_exc,         // (128, 400, 500) f32
    float* __restrict__ out_inh)         // (128, 400, 500) f32
{
    __shared__ __align__(16) _Float16 xsbuf[2 * SLAB];   // double-buffered x_t (51200 B)
    __shared__ __align__(16) _Float16 sh[16 * K2PAD];    // s_inh(t-1)         (13312 B)

    const int tid  = threadIdx.x;
    const int lane = tid & 31;
    const int wave = tid >> 5;
    const int half = lane >> 4;        // 0: lanes 0-15, 1: lanes 16-31
    const int lm   = lane & 15;
    const int b0   = blockIdx.x * 16;  // first batch row of this WG

    const int  nt0  = wave;            // 0..15
    const int  nt1  = wave + 16;       // 16..31
    const bool has1 = nt1 < NTILES;
    const int  nt1c = has1 ? nt1 : 0;  // clamped (loads stay in-bounds)

    // zero s_inh staging (incl. K padding, never rewritten)
    for (int c = tid; c < 16 * K2PAD; c += 512) sh[c] = (_Float16)0.0f;

    // persistent LIF state in WMMA C/D layout (2 N-tile slots per wave)
    v8f vexc[2], vinh[2], sprev[2], acc[2];
#pragma unroll
    for (int s = 0; s < 2; ++s)
#pragma unroll
        for (int i = 0; i < 8; ++i) { vexc[s][i] = 0.0f; vinh[s][i] = 0.0f; sprev[s][i] = 0.0f; acc[s][i] = 0.0f; }

    const v8h* sr = (const v8h*)(sh + lm * K2PAD + half * 8);   // s_inh A-frag base

    // LICM-opaque weight base pointers: rebound every t-iteration below so
    // the compiler cannot hoist 600+ VGPRs of weight fragments out of the
    // time loop (which spilled to scratch in the previous build).
    const _Float16* pwin_t = pwin;
    const _Float16* pwie_t = pwie;

    // ---- preload slab 0 ----
#if HAVE_TDM
    if (wave == 0)
        tdm_load_slab(xT + (size_t)b0 * K1PAD, (u32)(uintptr_t)&xsbuf[0]);
#else
    copy_slab(xT + (size_t)b0 * K1PAD, xsbuf, tid, 512);
#endif

    for (int t = 0; t < TSTEPS; ++t) {
        _Float16* xcur = xsbuf + (t & 1) * SLAB;

        // defeat LICM: weight pointers are "redefined" each step
        asm("" : "+s"(pwin_t), "+s"(pwie_t));

#if HAVE_TDM
        if (wave == 0) __builtin_amdgcn_s_wait_tensorcnt(0);
#endif
        __syncthreads();   // A: slab(t) ready in xcur; sh holds s_inh(t-1)

        // ---- kick off slab(t+1) into the other LDS buffer ----
        if (t + 1 < TSTEPS) {
            const _Float16* nsrc = xT + ((size_t)(t + 1) * BATCH + b0) * K1PAD;
            _Float16* nbuf = xsbuf + ((t + 1) & 1) * SLAB;
#if HAVE_TDM
            if (wave == 0) tdm_load_slab(nsrc, (u32)(uintptr_t)nbuf);
#else
            copy_slab(nsrc, nbuf, tid, 512);
#endif
        }
        // L2 prefetch of slab(t+2)
        if (wave == 1 && t + 2 < TSTEPS) {
            const char* p = (const char*)(xT + ((size_t)(t + 2) * BATCH + b0) * K1PAD);
            for (int c = lane; c < (SLAB * 2) / 64; c += 32)
                __builtin_prefetch(p + c * 64, 0, 0);
        }

        // ---- matmul phase: I_exc = x_t @ w_in - s_inh @ w_ie ----
        // Both accumulators computed unconditionally (straight-line WMMA,
        // two independent chains); slot-1 results are discarded in the LIF
        // phase for waves without a second tile.
        v8f a0, a1;
#pragma unroll
        for (int i = 0; i < 8; ++i) { a0[i] = 0.0f; a1[i] = 0.0f; }

        {   // input -> exc: K = 800, 25 k-tiles, pipelined one tile ahead
            const v8h*  xr  = (const v8h*)(xcur + lm * K1PAD + half * 8);
            const v16h* b0p = (const v16h*)pwin_t + ((size_t)nt0  * K1PAD + lane);
            const v16h* b1p = (const v16h*)pwin_t + ((size_t)nt1c * K1PAD + lane);
            v16h_u aC, aN;
            v16h b0C, b1C, b0N, b1N;
            aC.h[0] = xr[0]; aC.h[1] = xr[2];
            b0C = b0p[0]; b1C = b1p[0];
#pragma unroll
            for (int kt = 0; kt < K1PAD / 32; ++kt) {
                if (kt + 1 < K1PAD / 32) {
                    aN.h[0] = xr[(kt + 1) * 4];
                    aN.h[1] = xr[(kt + 1) * 4 + 2];
                    b0N = b0p[(size_t)(kt + 1) * 32];
                    b1N = b1p[(size_t)(kt + 1) * 32];
                }
                a0 = WMMA_F16(aC.v, b0C, a0);
                a1 = WMMA_F16(aC.v, b1C, a1);
                if (kt + 1 < K1PAD / 32) { aC = aN; b0C = b0N; b1C = b1N; }
            }
        }
        {   // inh -> exc (pre-negated): K = 416, 13 k-tiles, pipelined
            const v16h* b0p = (const v16h*)pwie_t + ((size_t)nt0  * K2PAD + lane);
            const v16h* b1p = (const v16h*)pwie_t + ((size_t)nt1c * K2PAD + lane);
            v16h_u aC, aN;
            v16h b0C, b1C, b0N, b1N;
            aC.h[0] = sr[0]; aC.h[1] = sr[2];
            b0C = b0p[0]; b1C = b1p[0];
#pragma unroll
            for (int kt = 0; kt < K2PAD / 32; ++kt) {
                if (kt + 1 < K2PAD / 32) {
                    aN.h[0] = sr[(kt + 1) * 4];
                    aN.h[1] = sr[(kt + 1) * 4 + 2];
                    b0N = b0p[(size_t)(kt + 1) * 32];
                    b1N = b1p[(size_t)(kt + 1) * 32];
                }
                a0 = WMMA_F16(aC.v, b0C, a0);
                a1 = WMMA_F16(aC.v, b1C, a1);
                if (kt + 1 < K2PAD / 32) { aC = aN; b0C = b0N; b1C = b1N; }
            }
        }
        acc[0] = a0; acc[1] = a1;
        __syncthreads();   // B: all reads of sh (s_inh(t-1)) complete

        // ---- LIF phase: update state, emit spikes, refresh sh ----
#pragma unroll
        for (int s = 0; s < 2; ++s) {
            if (s == 1 && !has1) break;
            const int nt = (s == 0) ? nt0 : nt1;
#pragma unroll
            for (int i = 0; i < 8; ++i) {
                const int m = i + half * 8;                 // C-layout row
                const int j = nt * 16 + lm;                 // exc neuron
                const size_t oidx = ((size_t)(b0 + m) * N_EXC + j) * TSTEPS + t;

                float ve = vexc[s][i] * DECAY + acc[s][i];
                float se = ve >= 1.0f ? 1.0f : 0.0f;
                vexc[s][i] = ve >= 1.0f ? 0.0f : ve;

                float vi = vinh[s][i] * DECAY + WEI * sprev[s][i];
                float si = vi >= 1.0f ? 1.0f : 0.0f;
                vinh[s][i] = vi >= 1.0f ? 0.0f : vi;

                sprev[s][i] = se;
                out_exc[oidx] = se;
                out_inh[oidx] = si;
                sh[m * K2PAD + j] = (_Float16)si;           // s_inh(t) for t+1
            }
        }
        __syncthreads();   // C: sh writes visible before next step's reads
    }
}

// ---------------------------------------------------------------------------
// Launch
// ---------------------------------------------------------------------------
extern "C" void kernel_launch(void* const* d_in, const int* in_sizes, int n_in,
                              void* d_out, int out_size, void* d_ws, size_t ws_size,
                              hipStream_t stream) {
    const float* in_spikes = (const float*)d_in[0];   // (128, 784, 500)
    const float* w_in      = (const float*)d_in[1];   // (784, 400)
    const float* w_ie      = (const float*)d_in[2];   // (400, 400)
    float* out = (float*)d_out;

    // workspace layout
    char* ws = (char*)d_ws;
    const size_t xT_bytes   = (size_t)TSTEPS * BATCH * K1PAD * sizeof(_Float16); // 102.4 MB
    const size_t pwin_bytes = (size_t)NTILES * K1PAD * 16 * sizeof(_Float16);    // 640 KB
    _Float16* xT   = (_Float16*)ws;
    _Float16* pwin = (_Float16*)(ws + xT_bytes);
    _Float16* pwie = (_Float16*)(ws + xT_bytes + pwin_bytes);

    (void)in_sizes; (void)n_in; (void)out_size; (void)ws_size;

    {   // pack weights (relu + mask + negate + f16 + WMMA-B tile layout)
        int n = NTILES * K1PAD * 16;
        pack_w_in<<<(n + 255) / 256, 256, 0, stream>>>(w_in, pwin);
    }
    {
        int n = NTILES * K2PAD * 16;
        pack_w_ie<<<(n + 255) / 256, 256, 0, stream>>>(w_ie, pwie);
    }
    {   // transpose + f16-convert spikes to (T, B, K1PAD)
        int n = BATCH * K1PAD;
        transpose_x<<<(n + 255) / 256, 256, 0, stream>>>(in_spikes, xT);
    }
    // main persistent scan: 8 WGs x 512 threads (one WG per 16 batch rows)
    float* out_exc = out;
    float* out_inh = out + (size_t)BATCH * N_EXC * TSTEPS;
    snn_scan<<<BATCH / 16, 512, 0, stream>>>(xT, pwin, pwie, out_exc, out_inh);
}
